// GAT_44452911513782
// MI455X (gfx1250) — matrix operand order repaired and checked by reading, not compiled, not run on hardware
//
#include <hip/hip_runtime.h>
#include <hip/hip_bf16.h>

typedef __bf16 bf16;
typedef __attribute__((ext_vector_type(16))) __bf16 v16bf;
typedef __attribute__((ext_vector_type(8)))  __bf16 v8bf;
typedef __attribute__((ext_vector_type(8)))  float  v8f;

#define NEG_SLOPE 0.2f

// ---------------------------------------------------------------- utilities

__global__ void fill_kernel(float* __restrict__ p, float v, int n) {
  int t = blockIdx.x * blockDim.x + threadIdx.x;
  if (t < n) p[t] = v;
}

__device__ __forceinline__ void atomicMaxF(float* addr, float val) {
  unsigned int* ua = (unsigned int*)addr;
  unsigned int old = __float_as_uint(*addr);
  while (__uint_as_float(old) < val) {
    unsigned int assumed = old;
    old = atomicCAS(ua, assumed, __float_as_uint(val));
    if (old == assumed) break;
  }
}

// ------------------------------------------------- WMMA fragment pre-packing
// Packed layout (bf16): elem[((tile * KC + kc) * 32 + lane) * 16 + h]
//   A (16xK row tiles):  row = tile*16 + (lane&15),
//                        K   = kc*32 + ((h&8)<<1) + ((lane>>4)<<3) + (h&7)
//   B (Kx16 col tiles):  col = tile*16 + (lane&15),
//                        K   = kc*32 + ((h&8)<<1) + ((lane>>4)<<3) + (h&7)
// This is exactly the per-lane VGPR order of V_WMMA_F32_16X16X32_BF16
// (ISA 7.12.2), so each lane's fragment is one contiguous 32B run.

__global__ void pack_a_kernel(const float* __restrict__ in,
                              bf16* __restrict__ out, int K, int kcShift,
                              long long total) {
  long long t = (long long)blockIdx.x * blockDim.x + threadIdx.x;
  if (t >= total) return;
  int h    = (int)(t & 15);
  int lane = (int)((t >> 4) & 31);
  long long rest = t >> 9;
  int kc = (int)(rest & ((1 << kcShift) - 1));
  long long mt = rest >> kcShift;
  int ln = lane & 15, g = lane >> 4;
  long long row = mt * 16 + ln;
  int kk = (kc << 5) + ((h & 8) << 1) + (g << 3) + (h & 7);
  out[t] = (bf16)in[row * K + kk];
}

__global__ void pack_b_kernel(const float* __restrict__ W,
                              bf16* __restrict__ out, int Nc, int kcShift,
                              int total) {
  int t = blockIdx.x * blockDim.x + threadIdx.x;
  if (t >= total) return;
  int h    = t & 15;
  int lane = (t >> 4) & 31;
  int rest = t >> 9;
  int kc = rest & ((1 << kcShift) - 1);
  int ct = rest >> kcShift;
  int ln = lane & 15, g = lane >> 4;
  int col = ct * 16 + ln;
  int kk = (kc << 5) + ((h & 8) << 1) + (g << 3) + (h & 7);
  out[t] = (bf16)W[(size_t)kk * Nc + col];
}

// ---------------------------------------------------------------- WMMA GEMM
// C[M x Nc] (f32) = A[M x K] * B[K x Nc], both pre-packed bf16 fragments.
// Block = 128 threads (4 wave32); each wave computes a 16x32 strip (two
// 16x16 WMMA tiles sharing the A fragment). grid = (M/16, Nc/128).
#define SHUF16(lo, hi)                                                        \
  __builtin_shufflevector(lo, hi, 0, 1, 2, 3, 4, 5, 6, 7, 8, 9, 10, 11, 12,  \
                          13, 14, 15)

__global__ __launch_bounds__(128) void gemm_packed_wmma_kernel(
    const bf16* __restrict__ Ap, const bf16* __restrict__ Bp,
    float* __restrict__ C, int KC, int Nc) {
  const int lane = threadIdx.x & 31;
  const int wave = threadIdx.x >> 5;
  const int g = lane >> 4, ln = lane & 15;
  const long long mt = blockIdx.x;
  const int ct0 = (blockIdx.y * 4 + wave) * 2;

  const v8bf* ap  = (const v8bf*)Ap + (mt * KC * 32 + lane) * 2;
  const v8bf* bp0 = (const v8bf*)Bp + ((long long)ct0 * KC * 32 + lane) * 2;
  const v8bf* bp1 = (const v8bf*)Bp + ((long long)(ct0 + 1) * KC * 32 + lane) * 2;

  v8f acc0 = {}, acc1 = {};
  for (int kc = 0; kc < KC; ++kc) {
    __builtin_prefetch((const void*)(ap + 128), 0, 1);  // 2 k-chunks ahead
    v8bf a0 = ap[0], a1 = ap[1];
    v8bf b00 = bp0[0], b01 = bp0[1];
    v8bf b10 = bp1[0], b11 = bp1[1];
    ap += 64; bp0 += 64; bp1 += 64;  // 512 halfs per k-chunk
    v16bf a  = SHUF16(a0, a1);
    v16bf b0 = SHUF16(b00, b01);
    v16bf b1 = SHUF16(b10, b11);
    acc0 = __builtin_amdgcn_wmma_f32_16x16x32_bf16(false, a, false, b0,
                                                   (short)0, acc0, false, false);
    acc1 = __builtin_amdgcn_wmma_f32_16x16x32_bf16(false, a, false, b1,
                                                   (short)0, acc1, false, false);
  }

  const long long row0 = mt * 16 + (g << 3);
  const int col0 = ct0 * 16 + ln;
#pragma unroll
  for (int r = 0; r < 8; ++r) {
    C[(row0 + r) * Nc + col0]      = acc0[r];
    C[(row0 + r) * Nc + col0 + 16] = acc1[r];
  }
}

// ---------------------------------------------------------------- GAT pieces

// alpha_{s,d}[n,h] = sum_c h[n,h,c] * a_{src,dst}[h,c]
__global__ void alpha_kernel(const float* __restrict__ h,
                             const float* __restrict__ a_s,
                             const float* __restrict__ a_d,
                             float* __restrict__ as_, float* __restrict__ ad_,
                             int N, int H, int C) {
  int t = blockIdx.x * blockDim.x + threadIdx.x;
  if (t >= N * H) return;
  int n = t / H, hh = t % H;
  const float* hp  = h + (size_t)n * H * C + (size_t)hh * C;
  const float* asp = a_s + hh * C;
  const float* adp = a_d + hh * C;
  float s = 0.f, d = 0.f;
  for (int c = 0; c < C; ++c) {
    float v = hp[c];
    s += v * asp[c];
    d += v * adp[c];
  }
  as_[t] = s;
  ad_[t] = d;
}

// Edge ids e < E come from edge_index; e >= E are self-loops (src=dst=e-E).
__device__ __forceinline__ void edge_sd(const int* ei, int E, int e, int& s,
                                        int& d) {
  if (e < E) { s = ei[e]; d = ei[E + e]; }
  else       { s = e - E; d = e - E; }
}

__global__ void edge_max_kernel(const int* __restrict__ ei, int E, int N,
                                const float* __restrict__ as_,
                                const float* __restrict__ ad_,
                                float* __restrict__ m_, int hShift) {
  int t = blockIdx.x * blockDim.x + threadIdx.x;
  int H = 1 << hShift;
  int Etot = E + N;
  if (t >= Etot * H) return;
  int e = t >> hShift, hh = t & (H - 1);
  int s, d;
  edge_sd(ei, E, e, s, d);
  float v = as_[s * H + hh] + ad_[d * H + hh];
  v = v > 0.f ? v : NEG_SLOPE * v;
  atomicMaxF(&m_[d * H + hh], v);
}

__global__ void edge_expsum_kernel(const int* __restrict__ ei, int E, int N,
                                   const float* __restrict__ as_,
                                   const float* __restrict__ ad_,
                                   const float* __restrict__ m_,
                                   float* __restrict__ eexp,
                                   float* __restrict__ den_, int hShift) {
  int t = blockIdx.x * blockDim.x + threadIdx.x;
  int H = 1 << hShift;
  int Etot = E + N;
  if (t >= Etot * H) return;
  int e = t >> hShift, hh = t & (H - 1);
  int s, d;
  edge_sd(ei, E, e, s, d);
  float v = as_[s * H + hh] + ad_[d * H + hh];
  v = v > 0.f ? v : NEG_SLOPE * v;
  float ee = __expf(v - m_[d * H + hh]);
  eexp[t] = ee;
  atomicAdd(&den_[d * H + hh], ee);
}

// Normalize in place: eexp[e,h] <- eexp[e,h] / (den[d,h] + eps)
__global__ void edge_alpha_kernel(const int* __restrict__ ei, int E, int N,
                                  const float* __restrict__ den_,
                                  float* __restrict__ eexp, int hShift) {
  int t = blockIdx.x * blockDim.x + threadIdx.x;
  int H = 1 << hShift;
  int Etot = E + N;
  if (t >= Etot * H) return;
  int e = t >> hShift, hh = t & (H - 1);
  int s, d;
  edge_sd(ei, E, e, s, d);
  (void)s;
  eexp[t] = eexp[t] / (den_[d * H + hh] + 1e-16f);
}

// out[d, f..f+3] += h[s, f..f+3] * alpha(e, head(f)); float4 per thread.
// Requires C % 4 == 0 (layers 1 & 2).
__global__ void edge_aggregate4_kernel(const int* __restrict__ ei, int E,
                                       int N, const float* __restrict__ hfeat,
                                       const float* __restrict__ alpha,
                                       float* __restrict__ out, int hcShift,
                                       int cShift) {
  unsigned int t = blockIdx.x * blockDim.x + threadIdx.x;
  unsigned int Q = 1u << (hcShift - 2);           // HC/4 quads per edge
  unsigned int Etot = (unsigned int)(E + N);
  if (t >= Etot * Q) return;
  unsigned int e = t >> (hcShift - 2);
  unsigned int f = (t & (Q - 1)) << 2;
  int hh = (int)(f >> cShift);
  int H = (int)(1u << (hcShift - cShift));
  unsigned int HC = 1u << hcShift;
  int s, d;
  edge_sd(ei, E, (int)e, s, d);
  float al = alpha[(size_t)e * H + hh];
  const float4 v = *(const float4*)&hfeat[(size_t)s * HC + f];
  float* o = &out[(size_t)d * HC + f];
  atomicAdd(o + 0, v.x * al);
  atomicAdd(o + 1, v.y * al);
  atomicAdd(o + 2, v.z * al);
  atomicAdd(o + 3, v.w * al);
}

// Scalar variant (layer 3, HC == 2).
__global__ void edge_aggregate_kernel(const int* __restrict__ ei, int E, int N,
                                      const float* __restrict__ hfeat,
                                      const float* __restrict__ alpha,
                                      float* __restrict__ out, int hcShift,
                                      int cShift) {
  unsigned int t = blockIdx.x * blockDim.x + threadIdx.x;
  unsigned int HC = 1u << hcShift;
  unsigned int Etot = (unsigned int)(E + N);
  if (t >= Etot * HC) return;
  unsigned int e = t >> hcShift;
  unsigned int f = t & (HC - 1);
  int hh = (int)(f >> cShift);
  int H = (int)(HC >> cShift);
  int s, d;
  edge_sd(ei, E, (int)e, s, d);
  float al = alpha[(size_t)e * H + hh];
  atomicAdd(&out[(size_t)d * HC + f], hfeat[(size_t)s * HC + f] * al);
}

// acc += bias; optional ELU (in place, f32).
__global__ void epilogue_kernel(float* __restrict__ acc,
                                const float* __restrict__ bias,
                                unsigned int total, unsigned int fMask,
                                int do_elu) {
  unsigned int t = blockIdx.x * blockDim.x + threadIdx.x;
  if (t >= total) return;
  float v = acc[t] + bias[t & fMask];
  if (do_elu) v = v > 0.f ? v : (__expf(v) - 1.f);
  acc[t] = v;
}

// Tiny final GEMM: [N x K] (f32) @ [K x Nc] (f32), Nc == 2.
__global__ void gemm3_kernel(const float* __restrict__ in,
                             const float* __restrict__ W,
                             float* __restrict__ out, int N, int K, int Nc) {
  int t = blockIdx.x * blockDim.x + threadIdx.x;
  if (t >= N * Nc) return;
  int n = t / Nc, j = t % Nc;
  float s = 0.f;
  for (int k = 0; k < K; ++k) s += in[(size_t)n * K + k] * W[k * Nc + j];
  out[t] = s;
}

__global__ void final_bias_kernel(const float* __restrict__ acc,
                                  const float* __restrict__ b,
                                  float* __restrict__ out, int total,
                                  int mask) {
  int t = blockIdx.x * blockDim.x + threadIdx.x;
  if (t < total) out[t] = acc[t] + b[t & mask];
}

// ---------------------------------------------------------------- launcher

extern "C" void kernel_launch(void* const* d_in, const int* in_sizes, int n_in,
                              void* d_out, int out_size, void* d_ws,
                              size_t ws_size, hipStream_t stream) {
  (void)n_in; (void)out_size; (void)ws_size;
  const float* x    = (const float*)d_in[0];
  const int*   ei   = (const int*)d_in[1];
  const float* W1   = (const float*)d_in[2];
  const float* a_s1 = (const float*)d_in[3];
  const float* a_d1 = (const float*)d_in[4];
  const float* b1   = (const float*)d_in[5];
  const float* W2   = (const float*)d_in[6];
  const float* a_s2 = (const float*)d_in[7];
  const float* a_d2 = (const float*)d_in[8];
  const float* b2   = (const float*)d_in[9];
  const float* W3   = (const float*)d_in[10];
  const float* a_s3 = (const float*)d_in[11];
  const float* a_d3 = (const float*)d_in[12];
  const float* b3   = (const float*)d_in[13];

  const int N = in_sizes[0] / 128;  // 100000 (divisible by 16)
  const int E = in_sizes[1] / 2;    // 400000
  const int Etot = E + N;           // with self-loops

  // Workspace layout (256B aligned blocks).
  char* ws = (char*)d_ws;
  size_t off = 0;
  auto alloc = [&](size_t bytes) -> char* {
    char* p = ws + off;
    off += (bytes + 255) & ~(size_t)255;
    return p;
  };
  bf16*  Ap   = (bf16*)alloc((size_t)N * 512 * 2);   // packed A fragments
  bf16*  Bp   = (bf16*)alloc((size_t)512 * 256 * 2); // packed B fragments
  float* hbuf = (float*)alloc((size_t)N * 512 * 4);  // h = x @ W
  float* obuf = (float*)alloc((size_t)N * 512 * 4);  // aggregated output
  float* as_  = (float*)alloc((size_t)N * 4 * 4);
  float* ad_  = (float*)alloc((size_t)N * 4 * 4);
  float* m_   = (float*)alloc((size_t)N * 4 * 4);
  float* den_ = (float*)alloc((size_t)N * 4 * 4);
  float* eexp = (float*)alloc((size_t)Etot * 4 * 4);
  float* h3   = (float*)alloc((size_t)N * 2 * 4);
  float* out3 = (float*)alloc((size_t)N * 2 * 4);

  dim3 blk(256);
  auto nblk = [](long long n) { return dim3((unsigned)((n + 255) / 256)); };

  // ---------------- Layer 1: 128 -> 4x128 concat ----------------
  pack_a_kernel<<<nblk((long long)N * 128), blk, 0, stream>>>(
      x, Ap, 128, 2, (long long)N * 128);
  pack_b_kernel<<<nblk(128 * 512), blk, 0, stream>>>(W1, Bp, 512, 2, 128 * 512);
  gemm_packed_wmma_kernel<<<dim3(N / 16, 512 / 128), 128, 0, stream>>>(
      Ap, Bp, hbuf, 128 / 32, 512);
  alpha_kernel<<<nblk(N * 4), blk, 0, stream>>>(hbuf, a_s1, a_d1, as_, ad_, N, 4, 128);
  fill_kernel<<<nblk(N * 4), blk, 0, stream>>>(m_, -1e30f, N * 4);
  fill_kernel<<<nblk(N * 4), blk, 0, stream>>>(den_, 0.f, N * 4);
  fill_kernel<<<nblk((long long)N * 512), blk, 0, stream>>>(obuf, 0.f, N * 512);
  edge_max_kernel<<<nblk((long long)Etot * 4), blk, 0, stream>>>(ei, E, N, as_, ad_, m_, 2);
  edge_expsum_kernel<<<nblk((long long)Etot * 4), blk, 0, stream>>>(
      ei, E, N, as_, ad_, m_, eexp, den_, 2);
  edge_alpha_kernel<<<nblk((long long)Etot * 4), blk, 0, stream>>>(ei, E, N, den_, eexp, 2);
  edge_aggregate4_kernel<<<nblk((long long)Etot * 128), blk, 0, stream>>>(
      ei, E, N, hbuf, eexp, obuf, 9, 7);
  epilogue_kernel<<<nblk((long long)N * 512), blk, 0, stream>>>(
      obuf, b1, (unsigned)(N * 512), 511u, 1);

  // ---------------- Layer 2: 512 -> 4x64 concat -----------------
  pack_a_kernel<<<nblk((long long)N * 512), blk, 0, stream>>>(
      obuf, Ap, 512, 4, (long long)N * 512);
  pack_b_kernel<<<nblk(512 * 256), blk, 0, stream>>>(W2, Bp, 256, 4, 512 * 256);
  gemm_packed_wmma_kernel<<<dim3(N / 16, 256 / 128), 128, 0, stream>>>(
      Ap, Bp, hbuf, 512 / 32, 256);
  alpha_kernel<<<nblk(N * 4), blk, 0, stream>>>(hbuf, a_s2, a_d2, as_, ad_, N, 4, 64);
  fill_kernel<<<nblk(N * 4), blk, 0, stream>>>(m_, -1e30f, N * 4);
  fill_kernel<<<nblk(N * 4), blk, 0, stream>>>(den_, 0.f, N * 4);
  fill_kernel<<<nblk((long long)N * 256), blk, 0, stream>>>(obuf, 0.f, N * 256);
  edge_max_kernel<<<nblk((long long)Etot * 4), blk, 0, stream>>>(ei, E, N, as_, ad_, m_, 2);
  edge_expsum_kernel<<<nblk((long long)Etot * 4), blk, 0, stream>>>(
      ei, E, N, as_, ad_, m_, eexp, den_, 2);
  edge_alpha_kernel<<<nblk((long long)Etot * 4), blk, 0, stream>>>(ei, E, N, den_, eexp, 2);
  edge_aggregate4_kernel<<<nblk((long long)Etot * 64), blk, 0, stream>>>(
      ei, E, N, hbuf, eexp, obuf, 8, 6);
  epilogue_kernel<<<nblk((long long)N * 256), blk, 0, stream>>>(
      obuf, b2, (unsigned)(N * 256), 255u, 1);

  // ---------------- Layer 3: 256 -> 2, heads=1, mean ------------
  gemm3_kernel<<<nblk(N * 2), blk, 0, stream>>>(obuf, W3, h3, N, 256, 2);
  alpha_kernel<<<nblk(N), blk, 0, stream>>>(h3, a_s3, a_d3, as_, ad_, N, 1, 2);
  fill_kernel<<<nblk(N), blk, 0, stream>>>(m_, -1e30f, N);
  fill_kernel<<<nblk(N), blk, 0, stream>>>(den_, 0.f, N);
  fill_kernel<<<nblk(N * 2), blk, 0, stream>>>(out3, 0.f, N * 2);
  edge_max_kernel<<<nblk(Etot), blk, 0, stream>>>(ei, E, N, as_, ad_, m_, 0);
  edge_expsum_kernel<<<nblk(Etot), blk, 0, stream>>>(ei, E, N, as_, ad_, m_, eexp, den_, 0);
  edge_alpha_kernel<<<nblk(Etot), blk, 0, stream>>>(ei, E, N, den_, eexp, 0);
  edge_aggregate_kernel<<<nblk((long long)Etot * 2), blk, 0, stream>>>(
      ei, E, N, h3, eexp, out3, 1, 1);
  final_bias_kernel<<<nblk(N * 2), blk, 0, stream>>>(out3, b3, (float*)d_out, N * 2, 1);
}
